// TransformerLayer_11690900979818
// MI455X (gfx1250) — matrix operand order repaired and checked by reading, not compiled, not run on hardware
//
#include <hip/hip_runtime.h>

// ---------------------------------------------------------------------------
// Types
// ---------------------------------------------------------------------------
typedef __bf16 bf16_t;
typedef __attribute__((ext_vector_type(16))) __bf16 v16bf;
typedef __attribute__((ext_vector_type(8)))  float  v8f;
typedef __attribute__((ext_vector_type(4)))  unsigned int u32x4;
typedef __attribute__((ext_vector_type(8)))  int i32x8;
typedef __attribute__((ext_vector_type(4)))  int i32x4;

union FragU {            // one WMMA 16-bit operand (16 bf16 per lane = 32B)
    v16bf v;
    uint4 u[2];
};
union V8U {              // 8 bf16 <-> uint4
    uint4  u;
    bf16_t e[8];
};
union V4U {              // 4 bf16 <-> uint2
    uint2  u;
    bf16_t e[4];
};

#define NEG_INF (-3.0e38f)

static __device__ __forceinline__ bf16_t f2bf(float f) {
    unsigned int u = __builtin_bit_cast(unsigned int, f);
    u += 0x7FFFu + ((u >> 16) & 1u);          // round-to-nearest-even
    unsigned short s = (unsigned short)(u >> 16);
    return __builtin_bit_cast(bf16_t, s);
}

static __device__ __forceinline__ v8f wmma_bf16(v16bf a, v16bf b, v8f c) {
    return __builtin_amdgcn_wmma_f32_16x16x32_bf16(
        /*neg_a=*/false, a, /*neg_b=*/false, b,
        /*c_mod=*/(short)0, c, /*reuse_a=*/false, /*reuse_b=*/false);
}

// ---------------------------------------------------------------------------
// Problem constants
// ---------------------------------------------------------------------------
#define DMODEL 1024
#define NHEAD  16
#define HD     64
#define SEQT   2048
#define NB     2
#define FDIM   4096
#define MT     (NB * SEQT)       // 4096 rows

// ---------------------------------------------------------------------------
// fp32 -> bf16 weight conversion (vectorized, one-shot per launch)
// ---------------------------------------------------------------------------
__global__ void cvt_f32_bf16_kernel(const float* __restrict__ src,
                                    bf16_t* __restrict__ dst, int n4) {
    const int i = blockIdx.x * blockDim.x + threadIdx.x;
    if (i < n4) {
        float4 f = ((const float4*)src)[i];
        V4U w;
        w.e[0] = f2bf(f.x); w.e[1] = f2bf(f.y);
        w.e[2] = f2bf(f.z); w.e[3] = f2bf(f.w);
        ((uint2*)dst)[i] = w.u;
    }
}

// ---------------------------------------------------------------------------
// LayerNorm: fp32 [rows, 1024] -> bf16 [rows, 1024]; 256 threads/row
// ---------------------------------------------------------------------------
__global__ void ln_bf16_kernel(const float* __restrict__ x,
                               const float* __restrict__ gamma,
                               const float* __restrict__ beta,
                               bf16_t* __restrict__ out) {
    __shared__ float red[256];
    const int row = blockIdx.x;
    const int tid = threadIdx.x;
    const float* xr = x + (size_t)row * DMODEL;

    float4 xv = ((const float4*)xr)[tid];              // elements tid*4..+3
    red[tid] = xv.x + xv.y + xv.z + xv.w;
    __syncthreads();
    for (int o = 128; o > 0; o >>= 1) {
        if (tid < o) red[tid] += red[tid + o];
        __syncthreads();
    }
    const float mu = red[0] * (1.0f / DMODEL);
    __syncthreads();

    float d0 = xv.x - mu, d1 = xv.y - mu, d2 = xv.z - mu, d3 = xv.w - mu;
    red[tid] = d0 * d0 + d1 * d1 + d2 * d2 + d3 * d3;
    __syncthreads();
    for (int o = 128; o > 0; o >>= 1) {
        if (tid < o) red[tid] += red[tid + o];
        __syncthreads();
    }
    const float rstd = rsqrtf(red[0] * (1.0f / DMODEL) + 1e-7f);

    const int c = tid * 4;
    bf16_t* orow = out + (size_t)row * DMODEL;
    orow[c + 0] = f2bf(d0 * rstd * gamma[c + 0] + beta[c + 0]);
    orow[c + 1] = f2bf(d1 * rstd * gamma[c + 1] + beta[c + 1]);
    orow[c + 2] = f2bf(d2 * rstd * gamma[c + 2] + beta[c + 2]);
    orow[c + 3] = f2bf(d3 * rstd * gamma[c + 3] + beta[c + 3]);
}

// ---------------------------------------------------------------------------
// Tiled WMMA GEMM: C[M,N] = act(A_bf16[M,K] @ W_bf16[K,N] + bias (+ resid))
// Block tile 128x128, BK=32, 256 threads = 8 waves.
// A tile staged by the Tensor Data Mover (TDM) with LDS padding 64B->80B rows;
// B tile transposed into LDS (col-major) by the waves while the DMA runs.
// ---------------------------------------------------------------------------
template <bool OBF16, bool RELU, bool RESID>
__global__ void gemm128_kernel(const bf16_t* __restrict__ A,
                               const bf16_t* __restrict__ Bw,
                               const float*  __restrict__ bias,
                               const float*  __restrict__ resid,
                               void* __restrict__ outv,
                               int M, int N, int K) {
    constexpr int BM = 128, BN = 128, BK = 32, LSTR = 40;  // 80B row pitch
    __shared__ __align__(16) bf16_t Asm[BM * LSTR];        // row-major  [m][k]
    __shared__ __align__(16) bf16_t Bsm[BN * LSTR];        // col-major  [n][k]

    const int tid  = threadIdx.x;
    const int lane = tid & 31;
    const int wave = tid >> 5;
    const int wm = wave & 3;      // 4 row groups of 32
    const int wn = wave >> 2;     // 2 col groups of 64
    const int l16 = lane & 15;
    const int lh  = lane >> 4;    // lane half

    const int m0 = blockIdx.y * BM;
    const int n0 = blockIdx.x * BN;

    // --- TDM descriptor group 1 (loop invariant): bf16 elements, 2D tile
    // 32(K) x 128(M), tensor dims KxM, row stride K, LDS pad 4 DW / 16 DW.
    i32x8 g1a;
    g1a[0] = (1 << 16) | (1 << 20) | (3 << 22) | (3 << 25);
    g1a[1] = (int)((K & 0xFFFF) << 16);                    // tensor_dim0 lo
    g1a[2] = (int)(((unsigned)K >> 16) | ((unsigned)(M & 0xFFFF) << 16));
    g1a[3] = (int)(((unsigned)M >> 16) | (32u << 16));     // tile_dim0 = 32
    g1a[4] = 128;                                          // tile_dim1 = 128
    g1a[5] = K;                                            // dim0 stride lo
    g1a[6] = 0;
    g1a[7] = 0;
    i32x4 gz;
    gz[0] = 0; gz[1] = 0; gz[2] = 0; gz[3] = 0;
    i32x8 gz8;
    gz8[0] = 0; gz8[1] = 0; gz8[2] = 0; gz8[3] = 0;
    gz8[4] = 0; gz8[5] = 0; gz8[6] = 0; gz8[7] = 0;
    const unsigned lds_a = (unsigned)(unsigned long long)(&Asm[0]);
    const unsigned long long ga_base =
        (unsigned long long)(const void*)(A + (size_t)m0 * K);

    v8f acc[2][4];
#pragma unroll
    for (int i = 0; i < 2; ++i)
#pragma unroll
        for (int j = 0; j < 4; ++j)
#pragma unroll
            for (int r = 0; r < 8; ++r) acc[i][j][r] = 0.0f;

    for (int kt = 0; kt < K; kt += BK) {
        __syncthreads();                 // previous tiles fully consumed
        // --- A tile via Tensor Data Mover (EXEC-independent, SGPR operands)
        if (tid == 0) {
            u32x4 g0;
            unsigned long long ga = ga_base + (unsigned long long)kt * 2ull;
            g0[0] = 1u;                              // count=1
            g0[1] = lds_a;                           // LDS dest
            g0[2] = (unsigned)ga;                    // global addr lo
            g0[3] = (unsigned)(ga >> 32) | (2u << 30);  // addr hi | type=2
            __builtin_amdgcn_tensor_load_to_lds(g0, g1a, gz, gz, gz8, 0);
        }
        {   // --- B tile: 32x128 bf16, transpose into Bsm[n][k] while DMA runs
            const int k = tid >> 3, seg = tid & 7;
            const bf16_t* src = Bw + (size_t)(kt + k) * N + n0 + seg * 16;
            V8U w0, w1;
            w0.u = ((const uint4*)src)[0];
            w1.u = ((const uint4*)src)[1];
            bf16_t* dst = Bsm + (size_t)(seg * 16) * LSTR + k;
#pragma unroll
            for (int j = 0; j < 8; ++j) dst[j * LSTR] = w0.e[j];
#pragma unroll
            for (int j = 0; j < 8; ++j) dst[(8 + j) * LSTR] = w1.e[j];
            if (kt + BK < K)             // prefetch next weight tile row
                __builtin_prefetch(src + (size_t)BK * N, 0, 0);
        }
        if (tid == 0) __builtin_amdgcn_s_wait_tensorcnt((short)0);
        __syncthreads();                 // publish TDM + DS writes

        FragU afr[2], bfr[4];
#pragma unroll
        for (int i = 0; i < 2; ++i) {
            const bf16_t* p = Asm + (wm * 32 + i * 16 + l16) * LSTR + lh * 8;
            afr[i].u[0] = *(const uint4*)p;          // K = lh*8 .. +7
            afr[i].u[1] = *(const uint4*)(p + 16);   // K = lh*8+16 .. +23
        }
#pragma unroll
        for (int j = 0; j < 4; ++j) {
            const bf16_t* p = Bsm + (wn * 64 + j * 16 + l16) * LSTR + lh * 8;
            bfr[j].u[0] = *(const uint4*)p;          // K = lh*8 .. +7
            bfr[j].u[1] = *(const uint4*)(p + 16);   // K = lh*8+16 .. +23
        }
#pragma unroll
        for (int i = 0; i < 2; ++i)
#pragma unroll
            for (int j = 0; j < 4; ++j)
                acc[i][j] = wmma_bf16(afr[i].v, bfr[j].v, acc[i][j]);
    }

    // Epilogue. C layout: VGPR r -> m = r + 8*lane_half, n = lane%16.
    float*  outf = (float*)outv;
    bf16_t* outb = (bf16_t*)outv;
#pragma unroll
    for (int i = 0; i < 2; ++i) {
#pragma unroll
        for (int j = 0; j < 4; ++j) {
            const int row0 = m0 + wm * 32 + i * 16 + lh * 8;
            const int col  = n0 + wn * 64 + j * 16 + l16;
            const float bc = bias[col];
#pragma unroll
            for (int r = 0; r < 8; ++r) {
                const int row = row0 + r;
                float v = acc[i][j][r] + bc;
                if (RELU)  v = fmaxf(v, 0.0f);
                if (RESID) v += resid[(size_t)row * N + col];
                if (OBF16) outb[(size_t)row * N + col] = f2bf(v);
                else       outf[(size_t)row * N + col] = v;
            }
        }
    }
}

// ---------------------------------------------------------------------------
// Flash-style causal attention. One wave (32 threads) per (b, h, 16-query
// tile). K-dim HD=64 -> 2 WMMAs per score tile; P@V over 32-key blocks.
// ---------------------------------------------------------------------------
__global__ void attn_kernel(const bf16_t* __restrict__ Q,
                            const bf16_t* __restrict__ Kmat,
                            const bf16_t* __restrict__ V,
                            bf16_t* __restrict__ O) {
    constexpr int VSTR = 40;   // key-pitch for transposed V tile (80B rows)
    constexpr int PSTR = 32;   // P row pitch (64B)
    __shared__ __align__(16) bf16_t Vt[HD * VSTR];   // [hd][key] 5.1 KB
    __shared__ __align__(16) bf16_t Pl[16 * PSTR];   // [m][k]    1.0 KB

    const int lane = threadIdx.x;
    const int l16 = lane & 15;
    const int lh  = lane >> 4;
    const int q0 = blockIdx.x * 16;
    const int h  = blockIdx.y;
    const int b  = blockIdx.z;
    const size_t base = (size_t)b * SEQT * DMODEL + (size_t)h * HD;

    // Q fragments (A-layout): row = q0 + lane%16, K base = lh*8 (+16)
    const bf16_t* qrow = Q + base + (size_t)(q0 + l16) * DMODEL + lh * 8;
    FragU qa0, qa1;
    qa0.u[0] = *(const uint4*)(qrow +  0);
    qa0.u[1] = *(const uint4*)(qrow + 16);
    qa1.u[0] = *(const uint4*)(qrow + 32);
    qa1.u[1] = *(const uint4*)(qrow + 48);

    float mrow[8], lrow[8];
    v8f accO[4];
#pragma unroll
    for (int r = 0; r < 8; ++r) { mrow[r] = NEG_INF; lrow[r] = 0.0f; }
#pragma unroll
    for (int t = 0; t < 4; ++t)
#pragma unroll
        for (int r = 0; r < 8; ++r) accO[t][r] = 0.0f;

    const int jend = q0 + 16;                 // keys <= q0+15 (causal)
    for (int j0 = 0; j0 < jend; j0 += 32) {
        __syncthreads();
        // Stage V[j0..j0+31][0..63] transposed into Vt[hd][key]
#pragma unroll
        for (int it = 0; it < 8; ++it) {
            const int idx = it * 32 + lane;          // 0..255
            const int key = idx >> 3, seg = idx & 7;
            const bf16_t* vp = V + base + (size_t)(j0 + key) * DMODEL + seg * 8;
            V8U w; w.u = *(const uint4*)vp;
#pragma unroll
            for (int jj = 0; jj < 8; ++jj)
                Vt[(seg * 8 + jj) * VSTR + key] = w.e[jj];
        }
        __syncthreads();

        // Scores: S = Q @ K^T (two 16-key column tiles)
        v8f s0, s1;
#pragma unroll
        for (int r = 0; r < 8; ++r) { s0[r] = 0.0f; s1[r] = 0.0f; }
        {
            const bf16_t* k0 = Kmat + base + (size_t)(j0 + l16) * DMODEL + lh * 8;
            const bf16_t* k1 = Kmat + base + (size_t)(j0 + 16 + l16) * DMODEL + lh * 8;
            FragU kb;
            kb.u[0] = *(const uint4*)(k0 +  0); kb.u[1] = *(const uint4*)(k0 + 16);
            s0 = wmma_bf16(qa0.v, kb.v, s0);
            kb.u[0] = *(const uint4*)(k0 + 32); kb.u[1] = *(const uint4*)(k0 + 48);
            s0 = wmma_bf16(qa1.v, kb.v, s0);
            kb.u[0] = *(const uint4*)(k1 +  0); kb.u[1] = *(const uint4*)(k1 + 16);
            s1 = wmma_bf16(qa0.v, kb.v, s1);
            kb.u[0] = *(const uint4*)(k1 + 32); kb.u[1] = *(const uint4*)(k1 + 48);
            s1 = wmma_bf16(qa1.v, kb.v, s1);
        }

        // Online softmax (row = q0 + r + 8*lh; cols = j0+l16, j0+16+l16)
        float p0[8], p1[8];
#pragma unroll
        for (int r = 0; r < 8; ++r) {
            const int rowg = q0 + r + lh * 8;
            float a = s0[r] * 0.125f;            // 1/sqrt(64)
            float c = s1[r] * 0.125f;
            if (j0 + l16 > rowg)      a = NEG_INF;
            if (j0 + 16 + l16 > rowg) c = NEG_INF;
            float mx = fmaxf(a, c);
            mx = fmaxf(mx, __shfl_xor(mx, 1, 32));
            mx = fmaxf(mx, __shfl_xor(mx, 2, 32));
            mx = fmaxf(mx, __shfl_xor(mx, 4, 32));
            mx = fmaxf(mx, __shfl_xor(mx, 8, 32));
            mx = fmaxf(mx, mrow[r]);
            const float e0 = __expf(a - mx);
            const float e1 = __expf(c - mx);
            p0[r] = e0; p1[r] = e1;
            float sum = e0 + e1;
            sum += __shfl_xor(sum, 1, 32);
            sum += __shfl_xor(sum, 2, 32);
            sum += __shfl_xor(sum, 4, 32);
            sum += __shfl_xor(sum, 8, 32);
            const float sc = __expf(mrow[r] - mx);
            lrow[r] = lrow[r] * sc + sum;
            mrow[r] = mx;
#pragma unroll
            for (int t = 0; t < 4; ++t) accO[t][r] *= sc;
        }

        // P (16x32) -> LDS -> A-fragment layout
#pragma unroll
        for (int r = 0; r < 8; ++r) {
            const int m = r + lh * 8;
            Pl[m * PSTR + l16]      = f2bf(p0[r]);
            Pl[m * PSTR + 16 + l16] = f2bf(p1[r]);
        }
        __syncthreads();
        FragU pa;
        {
            const bf16_t* pp = Pl + l16 * PSTR + lh * 8;
            pa.u[0] = *(const uint4*)pp;
            pa.u[1] = *(const uint4*)(pp + 16);
        }
        // accO += P @ V   (4 n-slices of 16 hd)
#pragma unroll
        for (int t = 0; t < 4; ++t) {
            const bf16_t* vp = Vt + (t * 16 + l16) * VSTR + lh * 8;
            FragU vb;
            vb.u[0] = *(const uint4*)vp;
            vb.u[1] = *(const uint4*)(vp + 16);
            accO[t] = wmma_bf16(pa.v, vb.v, accO[t]);
        }
    }

    // Normalize and store context (bf16, [B,T,D] with head offset)
#pragma unroll
    for (int t = 0; t < 4; ++t) {
#pragma unroll
        for (int r = 0; r < 8; ++r) {
            const int rowg = q0 + r + lh * 8;
            const float val = accO[t][r] / lrow[r];
            O[base + (size_t)rowg * DMODEL + t * 16 + l16] = f2bf(val);
        }
    }
}

// ---------------------------------------------------------------------------
// Launch
// ---------------------------------------------------------------------------
extern "C" void kernel_launch(void* const* d_in, const int* in_sizes, int n_in,
                              void* d_out, int out_size, void* d_ws, size_t ws_size,
                              hipStream_t stream) {
    (void)in_sizes; (void)n_in; (void)out_size; (void)ws_size;
    const float* x   = (const float*)d_in[0];
    const float* Wq  = (const float*)d_in[1];
    const float* bq  = (const float*)d_in[2];
    const float* Wk  = (const float*)d_in[3];
    const float* bk  = (const float*)d_in[4];
    const float* Wv  = (const float*)d_in[5];
    const float* bv  = (const float*)d_in[6];
    const float* Wo  = (const float*)d_in[7];
    const float* bo  = (const float*)d_in[8];
    const float* W1  = (const float*)d_in[9];
    const float* b1  = (const float*)d_in[10];
    const float* W2  = (const float*)d_in[11];
    const float* b2  = (const float*)d_in[12];
    const float* g1  = (const float*)d_in[13];
    const float* be1 = (const float*)d_in[14];
    const float* g2  = (const float*)d_in[15];
    const float* be2 = (const float*)d_in[16];

    char* ws = (char*)d_ws;
    const size_t MB = 1u << 20;
    bf16_t* xn1 = (bf16_t*)(ws + 0 * MB);    // 8 MB  [MT, D]
    bf16_t* qb  = (bf16_t*)(ws + 8 * MB);    // 8 MB
    bf16_t* kb  = (bf16_t*)(ws + 16 * MB);   // 8 MB
    bf16_t* vb  = (bf16_t*)(ws + 24 * MB);   // 8 MB
    bf16_t* ctx = (bf16_t*)(ws + 32 * MB);   // 8 MB
    float*  x1  = (float*)(ws + 40 * MB);    // 16 MB [MT, D] fp32
    bf16_t* xn2 = (bf16_t*)(ws + 56 * MB);   // 8 MB
    bf16_t* h1  = (bf16_t*)(ws + 64 * MB);   // 32 MB [MT, F]
    bf16_t* Wqb = (bf16_t*)(ws + 96 * MB);   // 2 MB each
    bf16_t* Wkb = (bf16_t*)(ws + 98 * MB);
    bf16_t* Wvb = (bf16_t*)(ws + 100 * MB);
    bf16_t* Wob = (bf16_t*)(ws + 102 * MB);
    bf16_t* W1b = (bf16_t*)(ws + 104 * MB);  // 8 MB
    bf16_t* W2b = (bf16_t*)(ws + 112 * MB);  // 8 MB
    float*  out = (float*)d_out;

    // 0) weights -> bf16 (single pass; weights are reused 32x per tile)
    const int nDD4 = DMODEL * DMODEL / 4, nDF4 = DMODEL * FDIM / 4;
    cvt_f32_bf16_kernel<<<(nDD4 + 255) / 256, 256, 0, stream>>>(Wq, Wqb, nDD4);
    cvt_f32_bf16_kernel<<<(nDD4 + 255) / 256, 256, 0, stream>>>(Wk, Wkb, nDD4);
    cvt_f32_bf16_kernel<<<(nDD4 + 255) / 256, 256, 0, stream>>>(Wv, Wvb, nDD4);
    cvt_f32_bf16_kernel<<<(nDD4 + 255) / 256, 256, 0, stream>>>(Wo, Wob, nDD4);
    cvt_f32_bf16_kernel<<<(nDF4 + 255) / 256, 256, 0, stream>>>(W1, W1b, nDF4);
    cvt_f32_bf16_kernel<<<(nDF4 + 255) / 256, 256, 0, stream>>>(W2, W2b, nDF4);

    // 1) ln1(x) -> xn1 (bf16)
    ln_bf16_kernel<<<MT, 256, 0, stream>>>(x, g1, be1, xn1);

    // 2) Q/K/V projections (bf16 out)
    dim3 gQKV(DMODEL / 128, MT / 128);
    gemm128_kernel<true, false, false><<<gQKV, 256, 0, stream>>>(xn1, Wqb, bq, nullptr, qb, MT, DMODEL, DMODEL);
    gemm128_kernel<true, false, false><<<gQKV, 256, 0, stream>>>(xn1, Wkb, bk, nullptr, kb, MT, DMODEL, DMODEL);
    gemm128_kernel<true, false, false><<<gQKV, 256, 0, stream>>>(xn1, Wvb, bv, nullptr, vb, MT, DMODEL, DMODEL);

    // 3) causal attention -> ctx (bf16)
    attn_kernel<<<dim3(SEQT / 16, NHEAD, NB), 32, 0, stream>>>(qb, kb, vb, ctx);

    // 4) O projection + residual -> x1 (fp32)
    gemm128_kernel<false, false, true><<<gQKV, 256, 0, stream>>>(ctx, Wob, bo, x, x1, MT, DMODEL, DMODEL);

    // 5) ln2(x1) -> xn2 (bf16)
    ln_bf16_kernel<<<MT, 256, 0, stream>>>(x1, g2, be2, xn2);

    // 6) MLP1 (+ReLU) -> h1 (bf16)
    gemm128_kernel<true, true, false><<<dim3(FDIM / 128, MT / 128), 256, 0, stream>>>(xn2, W1b, b1, nullptr, h1, MT, FDIM, DMODEL);

    // 7) MLP2 + residual -> out (fp32)
    gemm128_kernel<false, false, true><<<dim3(DMODEL / 128, MT / 128), 256, 0, stream>>>(h1, W2b, b2, x1, out, MT, DMODEL, FDIM);
}
